// Strategist_64570538328619
// MI455X (gfx1250) — 8-bit matrix operand order repaired and checked by reading, compiled, not run on hardware
//
#include <hip/hip_runtime.h>
#include <math.h>


#define DIMW 512
#define K1   1536
#define HID  2048
#define NTOK 41472
#define TPB  16
#define EPSQ 1e-5f

typedef int v8i __attribute__((ext_vector_type(8)));

__device__ __forceinline__ v8i wmma_iu8(v8i a, v8i b, v8i c) {
  return __builtin_amdgcn_wmma_i32_16x16x64_iu8(true, a, true, b, c, false, false);
}

__device__ __forceinline__ v8i load_a_frag(const char* base, int pitch, int lane) {
  const char* p = base + (lane & 15) * pitch + (lane >> 4) * 8;
  v8i a;
  a[0] = *(const int*)(p +  0); a[1] = *(const int*)(p +  4);
  a[2] = *(const int*)(p + 16); a[3] = *(const int*)(p + 20);
  a[4] = *(const int*)(p + 32); a[5] = *(const int*)(p + 36);
  a[6] = *(const int*)(p + 48); a[7] = *(const int*)(p + 52);
  return a;
}

__device__ __forceinline__ v8i load_b_frag(const signed char* W, int Kdim, int ncol, int lane) {
  const signed char* p = W + (size_t)(ncol + (lane & 15)) * Kdim + (lane >> 4) * 8;
  v8i b;
  b[0] = *(const int*)(p +  0); b[1] = *(const int*)(p +  4);
  b[2] = *(const int*)(p + 16); b[3] = *(const int*)(p + 20);
  b[4] = *(const int*)(p + 32); b[5] = *(const int*)(p + 36);
  b[6] = *(const int*)(p + 48); b[7] = *(const int*)(p + 52);
  return b;
}

__device__ __forceinline__ float wave_max32(float v) {
#pragma unroll
  for (int m = 1; m < 32; m <<= 1) v = fmaxf(v, __shfl_xor(v, m, 32));
  return v;
}

__device__ __forceinline__ void quant_row512(const float* src, char* dst,
                                             float* scale_out, int t, int lane) {
  float vals[16]; float mx = 0.f;
#pragma unroll
  for (int j = 0; j < 16; ++j) { float v = src[lane + 32*j]; vals[j] = v; mx = fmaxf(mx, fabsf(v)); }
  mx = fmaxf(wave_max32(mx), EPSQ);
  float qs = 127.f / mx;
  if (lane == 0) scale_out[t] = mx * (1.f/127.f);
#pragma unroll
  for (int j = 0; j < 16; ++j) {
    int q = __float2int_rn(vals[j] * qs);
    q = q > 127 ? 127 : (q < -128 ? -128 : q);
    dst[lane + 32*j] = (char)q;
  }
}

__global__ void wq_reduce(const float* Win, const float* Wf1, const float* Wf2,
                          const float* Wz, const float* Wm, float* wscales) {
  __shared__ float red[256];
  const float* ptrs[5]  = { Win, Wf1, Wf2, Wz, Wm };
  const int    sizes[5] = { DIMW*K1, HID*DIMW, DIMW*HID, DIMW*DIMW, DIMW };
  const int t = blockIdx.x;
  const float* p = ptrs[t];
  const int n = sizes[t];
  float s = 0.f;
  for (int i = threadIdx.x; i < n; i += 256) s += fabsf(p[i]);
  red[threadIdx.x] = s;
  __syncthreads();
  for (int o = 128; o > 0; o >>= 1) {
    if (threadIdx.x < o) red[threadIdx.x] += red[threadIdx.x + o];
    __syncthreads();
  }
  if (threadIdx.x == 0) wscales[t] = fmaxf(red[0] / (float)n, EPSQ);
}

__global__ void wq_quant(const float* __restrict__ W, const float* __restrict__ wscales,
                         int idx, signed char* __restrict__ out, int n) {
  int i = blockIdx.x * 256 + threadIdx.x;
  if (i >= n) return;
  float inv = 1.f / wscales[idx];
  int q = __float2int_rn(W[i] * inv);
  q = q > 1 ? 1 : (q < -1 ? -1 : q);
  out[i] = (signed char)q;
}

__global__ void __launch_bounds__(256)
bitnet_fused(const float* __restrict__ x, const float* __restrict__ y,
             const float* __restrict__ sp,
             const signed char* __restrict__ Wq_in, const signed char* __restrict__ Wq_f1,
             const signed char* __restrict__ Wq_f2, const signed char* __restrict__ Wq_z,
             const signed char* __restrict__ Wq_m, const float* __restrict__ wsc,
             float* __restrict__ out_z, float* __restrict__ out_m) {
  extern __shared__ char smem[];
  char*  R0   = smem;
  float* Hbuf = (float*)(smem + 32768);
  char*  A8s  = smem + 65536;
  float* asc  = (float*)(smem + 73728);
  float* s2   = asc + 16;
  float* s3   = s2 + 16;
  float* wred = s3 + 16;
  int*   msum = (int*)(wred + 128);

  const int tid  = threadIdx.x;
  const int wid  = tid >> 5;
  const int lane = tid & 31;
  const size_t tokbase = (size_t)blockIdx.x * TPB;

  const float wsc_in = wsc[0], wsc_f1 = wsc[1], wsc_f2 = wsc[2],
              wsc_z = wsc[3],  wsc_m = wsc[4];

#pragma unroll
  for (int tt = 0; tt < 2; ++tt) {
    const int t = wid * 2 + tt;
    const size_t tok = tokbase + t;
    const float* px = x  + tok * DIMW;
    const float* py = y  + tok * DIMW;
    const float* ps = sp + tok * DIMW;
    float vals[48]; float mx = 0.f;
#pragma unroll
    for (int j = 0; j < 16; ++j) { float v = px[lane + 32*j]; vals[j]      = v; mx = fmaxf(mx, fabsf(v)); }
#pragma unroll
    for (int j = 0; j < 16; ++j) { float v = py[lane + 32*j]; vals[16 + j] = v; mx = fmaxf(mx, fabsf(v)); }
#pragma unroll
    for (int j = 0; j < 16; ++j) { float v = ps[lane + 32*j]; vals[32 + j] = v; mx = fmaxf(mx, fabsf(v)); }
    mx = fmaxf(wave_max32(mx), EPSQ);
    float qs = 127.f / mx;
    if (lane == 0) asc[t] = mx * (1.f/127.f);
    char* row = R0 + t * K1;
#pragma unroll
    for (int j = 0; j < 48; ++j) {
      int q = __float2int_rn(vals[j] * qs);
      q = q > 127 ? 127 : (q < -128 ? -128 : q);
      row[lane + 32*j] = (char)q;
    }
  }
  __syncthreads();

#pragma unroll
  for (int tl = 0; tl < 4; ++tl) {
    const int ncol = (wid * 4 + tl) * 16;
    v8i acc = {0,0,0,0,0,0,0,0};
    for (int ks = 0; ks < K1/64; ++ks) {
      __builtin_prefetch(Wq_in + (size_t)ncol * K1 + ks*64 + 128, 0, 0);
      v8i a = load_a_frag(R0 + ks*64, K1, lane);
      v8i b = load_b_frag(Wq_in + ks*64, K1, ncol, lane);
      acc = wmma_iu8(a, b, acc);
    }
    const int n = ncol + (lane & 15);
#pragma unroll
    for (int r = 0; r < 8; ++r) {
      const int m = r + 8 * (lane >> 4);
      Hbuf[m * DIMW + n] = (float)acc[r] * asc[m] * wsc_in;
    }
  }
  __syncthreads();

#pragma unroll
  for (int tt = 0; tt < 2; ++tt) {
    const int t = wid * 2 + tt;
    quant_row512(Hbuf + t * DIMW, A8s + t * DIMW, s2, t, lane);
  }
  __syncthreads();

  float ga[16][8];
  float pm[8];
#pragma unroll
  for (int r = 0; r < 8; ++r) pm[r] = 0.f;
#pragma unroll
  for (int tl = 0; tl < 16; ++tl) {
    const int ncol = (wid * 16 + tl) * 16;
    v8i acc = {0,0,0,0,0,0,0,0};
#pragma unroll
    for (int ks = 0; ks < DIMW/64; ++ks) {
      v8i a = load_a_frag(A8s + ks*64, DIMW, lane);
      v8i b = load_b_frag(Wq_f1 + ks*64, DIMW, ncol, lane);
      acc = wmma_iu8(a, b, acc);
    }
#pragma unroll
    for (int r = 0; r < 8; ++r) {
      const int m = r + 8 * (lane >> 4);
      float g = (float)acc[r] * s2[m] * wsc_f1;
      g = 0.5f * g * (1.f + erff(g * 0.70710678118f));
      ga[tl][r] = g;
      pm[r] = fmaxf(pm[r], fabsf(g));
    }
  }
#pragma unroll
  for (int r = 0; r < 8; ++r) {
#pragma unroll
    for (int mk = 1; mk < 16; mk <<= 1) pm[r] = fmaxf(pm[r], __shfl_xor(pm[r], mk, 32));
  }
  if ((lane & 15) == 0) {
#pragma unroll
    for (int r = 0; r < 8; ++r) wred[wid * 16 + r + 8 * (lane >> 4)] = pm[r];
  }
  __syncthreads();
  if (tid < 16) {
    float mx = 0.f;
#pragma unroll
    for (int w = 0; w < 8; ++w) mx = fmaxf(mx, wred[w * 16 + tid]);
    mx = fmaxf(mx, EPSQ);
    s2[tid]   = mx * (1.f/127.f);
    wred[tid] = 127.f / mx;
    msum[tid] = 0;
  }
  __syncthreads();
#pragma unroll
  for (int tl = 0; tl < 16; ++tl) {
    const int n = (wid * 16 + tl) * 16 + (lane & 15);
#pragma unroll
    for (int r = 0; r < 8; ++r) {
      const int m = r + 8 * (lane >> 4);
      int q = __float2int_rn(ga[tl][r] * wred[m]);
      q = q > 127 ? 127 : (q < -128 ? -128 : q);
      R0[m * HID + n] = (char)q;
    }
  }
  __syncthreads();

#pragma unroll
  for (int tl = 0; tl < 4; ++tl) {
    const int ncol = (wid * 4 + tl) * 16;
    v8i acc = {0,0,0,0,0,0,0,0};
    for (int ks = 0; ks < HID/64; ++ks) {
      v8i a = load_a_frag(R0 + ks*64, HID, lane);
      v8i b = load_b_frag(Wq_f2 + ks*64, HID, ncol, lane);
      acc = wmma_iu8(a, b, acc);
    }
    const int n = ncol + (lane & 15);
#pragma unroll
    for (int r = 0; r < 8; ++r) {
      const int m = r + 8 * (lane >> 4);
      Hbuf[m * DIMW + n] += (float)acc[r] * s2[m] * wsc_f2;
    }
  }
  __syncthreads();

#pragma unroll
  for (int tt = 0; tt < 2; ++tt) {
    const int t = wid * 2 + tt;
    quant_row512(Hbuf + t * DIMW, A8s + t * DIMW, s3, t, lane);
  }
  __syncthreads();

#pragma unroll
  for (int tl = 0; tl < 4; ++tl) {
    const int ncol = (wid * 4 + tl) * 16;
    v8i acc = {0,0,0,0,0,0,0,0};
#pragma unroll
    for (int ks = 0; ks < DIMW/64; ++ks) {
      v8i a = load_a_frag(A8s + ks*64, DIMW, lane);
      v8i b = load_b_frag(Wq_z + ks*64, DIMW, ncol, lane);
      acc = wmma_iu8(a, b, acc);
    }
    const int n = ncol + (lane & 15);
#pragma unroll
    for (int r = 0; r < 8; ++r) {
      const int m = r + 8 * (lane >> 4);
      out_z[(tokbase + m) * DIMW + n] = (float)acc[r] * s3[m] * wsc_z;
    }
  }

  {
    const int m = tid >> 4, seg = tid & 15;
    const signed char* ar = (const signed char*)(A8s + m * DIMW + seg * 32);
    const signed char* wr = Wq_m + seg * 32;
    int s = 0;
#pragma unroll
    for (int i = 0; i < 32; ++i) s += (int)ar[i] * (int)wr[i];
    atomicAdd(&msum[m], s);
  }
  __syncthreads();
  if (tid < 16) {
    float v = (float)msum[tid] * s3[tid] * wsc_m;
    out_m[tokbase + tid] = 1.f / (1.f + __expf(-v));
  }
}

extern "C" void kernel_launch(void* const* d_in, const int* in_sizes, int n_in,
                              void* d_out, int out_size, void* d_ws, size_t ws_size,
                              hipStream_t stream) {
  const float* x   = (const float*)d_in[0];
  const float* y   = (const float*)d_in[1];
  const float* sp  = (const float*)d_in[2];
  const float* Win = (const float*)d_in[3];
  const float* Wf1 = (const float*)d_in[4];
  const float* Wf2 = (const float*)d_in[5];
  const float* Wz  = (const float*)d_in[6];
  const float* Wm  = (const float*)d_in[7];
  (void)in_sizes; (void)n_in; (void)out_size; (void)ws_size;

  char* ws = (char*)d_ws;
  float*       wscales = (float*)ws;
  signed char* q_in = (signed char*)(ws + 256);
  signed char* q_f1 = q_in + (size_t)DIMW * K1;
  signed char* q_f2 = q_f1 + (size_t)HID * DIMW;
  signed char* q_z  = q_f2 + (size_t)DIMW * HID;
  signed char* q_m  = q_z  + (size_t)DIMW * DIMW;

  wq_reduce<<<5, 256, 0, stream>>>(Win, Wf1, Wf2, Wz, Wm, wscales);
  wq_quant<<<(DIMW*K1   + 255)/256, 256, 0, stream>>>(Win, wscales, 0, q_in, DIMW*K1);
  wq_quant<<<(HID*DIMW  + 255)/256, 256, 0, stream>>>(Wf1, wscales, 1, q_f1, HID*DIMW);
  wq_quant<<<(DIMW*HID  + 255)/256, 256, 0, stream>>>(Wf2, wscales, 2, q_f2, DIMW*HID);
  wq_quant<<<(DIMW*DIMW + 255)/256, 256, 0, stream>>>(Wz,  wscales, 3, q_z,  DIMW*DIMW);
  wq_quant<<<(DIMW      + 255)/256, 256, 0, stream>>>(Wm,  wscales, 4, q_m,  DIMW);

  float* out_z = (float*)d_out;
  float* out_m = out_z + (size_t)NTOK * DIMW;
  const size_t smem_bytes = 74496;
  bitnet_fused<<<NTOK / TPB, 256, smem_bytes, stream>>>(
      x, y, sp, q_in, q_f1, q_f2, q_z, q_m, wscales, out_z, out_m);
}
